// LPCDARTSSearchSpace_15247133900964
// MI455X (gfx1250) — compile-verified, matrix operand order checked by reading, not run on hardware
//
#include <hip/hip_runtime.h>
#include <cstdint>

typedef __attribute__((ext_vector_type(16))) __bf16 v16bf;
typedef __attribute__((ext_vector_type(8)))  float  v8f;

#define NPIX 65536            // 64 images * 32*32
#define PCH 64                // P
#define STATE_ELEMS (64*256*1024)
#define STATE_BYTES ((size_t)STATE_ELEMS*4)
#define ASTR 36               // LDS row stride (halves): 72B = 18 banks -> conflict-free
#define BSTR 36

// ---------------- small helper kernels ----------------

__global__ void softmax_kernel(const float* __restrict__ ap,
                               const float* __restrict__ norms,
                               float* __restrict__ sw) {
  int e = (int)threadIdx.x;
  if (e >= 10) return;
  int rem = e, node = 0;
  for (node = 0; node < 4; ++node) {
    if (rem <= node) break;
    rem -= node + 1;
  }
  int i = rem;
  float norm = fmaxf(norms[node*5 + i], 1e-5f);
  const float* a = ap + (node*5 + i)*12;
  float m = -1e30f;
  for (int j = 0; j < 12; ++j) m = fmaxf(m, a[j]/norm);
  float ex[12]; float s = 0.f;
  for (int j = 0; j < 12; ++j) { ex[j] = __expf(a[j]/norm - m); s += ex[j]; }
  for (int j = 0; j < 12; ++j) sw[e*12 + j] = ex[j] / s;
}

__global__ void zero_kernel(float* __restrict__ dst, int n) {
  int i = blockIdx.x*256 + threadIdx.x;
  if (i < n) dst[i] = 0.f;
}

__global__ void cvt_bf16_kernel(const float* __restrict__ src,
                                __bf16* __restrict__ dst, int n) {
  int i = blockIdx.x*256 + threadIdx.x;
  if (i < n) dst[i] = (__bf16)src[i];
}

// gather P channels of state -> bf16 xs[c][pix], and init p = w0*xs
__global__ void gather_kernel(const float* __restrict__ s,
                              const int* __restrict__ ids,
                              const float* __restrict__ swe,
                              __bf16* __restrict__ xsb,
                              float* __restrict__ p) {
  int i = blockIdx.x*256 + threadIdx.x;        // over PCH*NPIX
  if (i >= PCH*NPIX) return;
  int c   = i >> 16;
  int pix = i & (NPIX-1);
  int n   = pix >> 10;
  int hw  = pix & 1023;
  float v = s[(size_t)n*(256*1024) + (size_t)ids[c]*1024 + hw];
  xsb[i] = (__bf16)v;
  p[i]   = swe[0] * v;
}

// depthwise KxK conv (bf16 in/out), weights fp32 [c][K*K]
template<int K>
__global__ void dw_kernel(const __bf16* __restrict__ xs,
                          const float* __restrict__ dwW,
                          __bf16* __restrict__ out) {
  int i = blockIdx.x*256 + threadIdx.x;        // over PCH*NPIX
  if (i >= PCH*NPIX) return;
  int c   = i >> 16;
  int pix = i & (NPIX-1);
  int n = pix >> 10, h = (pix >> 5) & 31, w = pix & 31;
  const float* wk = dwW + c*K*K;
  float acc = 0.f;
  #pragma unroll
  for (int kh = 0; kh < K; ++kh) {
    int ih = h + kh - K/2;
    if ((unsigned)ih >= 32u) continue;
    #pragma unroll
    for (int kw = 0; kw < K; ++kw) {
      int iw = w + kw - K/2;
      if ((unsigned)iw >= 32u) continue;
      acc += (float)xs[(c << 16) + (n << 10) + (ih << 5) + iw] * wk[kh*K + kw];
    }
  }
  out[i] = (__bf16)acc;
}

// max3x3 + avg3x3 over full 256-ch state, accumulate into acc
__global__ void pool_kernel(const float* __restrict__ s,
                            const float* __restrict__ swe,
                            float* __restrict__ acc) {
  int i = blockIdx.x*256 + threadIdx.x;        // over STATE_ELEMS
  if (i >= STATE_ELEMS) return;
  int hw = i & 1023;
  int h = hw >> 5, w = hw & 31;
  int base = i - hw;
  float mx = -1e30f, sum = 0.f;
  #pragma unroll
  for (int dh = -1; dh <= 1; ++dh) {
    int ih = h + dh;
    if ((unsigned)ih >= 32u) continue;
    #pragma unroll
    for (int dw = -1; dw <= 1; ++dw) {
      int iw = w + dw;
      if ((unsigned)iw >= 32u) continue;
      float v = s[base + (ih << 5) + iw];
      mx = fmaxf(mx, v);
      sum += v;
    }
  }
  acc[i] += swe[5]*mx + swe[6]*(sum*(1.f/9.f));
}

// acc[n][ids[c]][hw] += p[c][pix]
__global__ void scatter_kernel(const float* __restrict__ p,
                               const int* __restrict__ ids,
                               float* __restrict__ acc) {
  int i = blockIdx.x*256 + threadIdx.x;        // over PCH*NPIX
  if (i >= PCH*NPIX) return;
  int c   = i >> 16;
  int pix = i & (NPIX-1);
  int n = pix >> 10, hw = pix & 1023;
  acc[(size_t)n*(256*1024) + (size_t)ids[c]*1024 + hw] += p[i];
}

// ---------------- WMMA implicit-GEMM conv (LDS-staged) ----------------
// xs: bf16 [64][NPIX] channel-major gathered activations
// wt: bf16 [64][64*K*K] oc-major, k contiguous
// p += (*swp) * conv(xs, wt)
// block = 256 threads = 8 waves -> 64 oc x 32 pix tile (one full image row).
// Per 32-k step: A tile staged LDS via async global->LDS b128 (rows padded to ASTR),
// B tile (im2col) staged via coalesced gathers + b64 LDS stores.
template<int K, int DIL>
__global__ __launch_bounds__(256)
void conv_wmma_kernel(const __bf16* __restrict__ xs,
                      const __bf16* __restrict__ wt,
                      const float* __restrict__ swp,
                      float* __restrict__ p) {
  constexpr int KK   = K*K;
  constexpr int Ktot = PCH*KK;               // multiple of 32 for K in {1,3,5,7}
  constexpr int PAD  = (K/2)*DIL;

  __shared__ __bf16 Asm[64*ASTR];            // 64 oc x 32 k (padded)
  __shared__ __bf16 Bsm[32*BSTR];            // 32 pix x 32 k (padded, k contiguous)

  const int tid  = threadIdx.x;
  const int lane = tid & 31;
  const int wave = tid >> 5;                 // 0..7
  const int half = lane >> 4;                // ISA K-half selector
  const int l    = lane & 15;
  const int ocTile = wave >> 1;              // 0..3
  const int pixSub = wave & 1;               // 0..1
  const int rowId  = blockIdx.x;             // (n,h) image row
  const int n = rowId >> 5;
  const int h = rowId & 31;
  const int pix = (rowId << 5) + pixSub*16 + l;
  const int oc  = ocTile*16 + l;

  // staging roles
  const int sA_oc  = tid >> 2;               // 0..63
  const int sA_seg = (tid & 3) * 8;          // 8-half segment within k-step
  const unsigned sA_lds = (unsigned)(uintptr_t)(&Asm[sA_oc*ASTR + sA_seg]);
  const int sB_col = tid & 31;               // pixel column within row
  const int sB_k0  = (tid >> 5) * 4;         // 4 k values per thread

  v8f acc = {};
  for (int kBase = 0; kBase < Ktot; kBase += 32) {
    __syncthreads();                         // protect LDS from previous compute
    // ---- stage A: async global -> LDS, 16B per thread ----
    {
      const __bf16* gp = wt + sA_oc*Ktot + kBase + sA_seg;
      asm volatile("global_load_async_to_lds_b128 %0, %1, off"
                   :: "v"(sA_lds), "v"((unsigned long long)(uintptr_t)gp)
                   : "memory");
    }
    // ---- stage B: im2col gather (coalesced in col), packed b64 store ----
    {
      union { __bf16 hh[4]; uint2 u; } pk;
      #pragma unroll
      for (int j = 0; j < 4; ++j) {
        int k  = kBase + sB_k0 + j;
        int ic = k / KK;
        int r  = k - ic*KK;
        int kh = r / K;
        int kw = r - kh*K;
        int ih = h + kh*DIL - PAD;
        int iw = sB_col + kw*DIL - PAD;
        __bf16 v = (__bf16)0.f;
        if ((unsigned)ih < 32u && (unsigned)iw < 32u)
          v = xs[(ic << 16) + (n << 10) + (ih << 5) + iw];
        pk.hh[j] = v;
      }
      *(uint2*)(&Bsm[sB_col*BSTR + sB_k0]) = pk.u;
    }
    asm volatile("s_wait_asynccnt 0" ::: "memory");
    __syncthreads();
    // ---- build fragments from LDS (vectorizable ds loads) ----
    v16bf a, b;
    const __bf16* ap = &Asm[oc*ASTR + half*8];
    const __bf16* bp = &Bsm[(pixSub*16 + l)*BSTR + half*8];
    #pragma unroll
    for (int t = 0; t < 8; ++t) {
      a[t] = ap[t];  a[t+8] = ap[16+t];
      b[t] = bp[t];  b[t+8] = bp[16+t];
    }
    acc = __builtin_amdgcn_wmma_f32_16x16x32_bf16(
        false, a, false, b, (short)0, acc, false, false);
  }
  const float wsc = *swp;
  #pragma unroll
  for (int r = 0; r < 8; ++r) {
    int ocr = ocTile*16 + r + 8*half;        // C/D layout: lanes 16-31 -> M+8
    p[(ocr << 16) + pix] += wsc * acc[r];
  }
}

// ---------------- host orchestration ----------------

static inline void run_conv_helper(int K, int DIL,
                                   const __bf16* xs, const __bf16* wt,
                                   const float* swp, float* p, hipStream_t st) {
  dim3 grid(NPIX/32), blk(256);
  if (K == 1)              conv_wmma_kernel<1,1><<<grid, blk, 0, st>>>(xs, wt, swp, p);
  else if (K==3 && DIL==1) conv_wmma_kernel<3,1><<<grid, blk, 0, st>>>(xs, wt, swp, p);
  else if (K==5 && DIL==1) conv_wmma_kernel<5,1><<<grid, blk, 0, st>>>(xs, wt, swp, p);
  else if (K==7 && DIL==1) conv_wmma_kernel<7,1><<<grid, blk, 0, st>>>(xs, wt, swp, p);
  else if (K==3 && DIL==2) conv_wmma_kernel<3,2><<<grid, blk, 0, st>>>(xs, wt, swp, p);
  else                     conv_wmma_kernel<5,2><<<grid, blk, 0, st>>>(xs, wt, swp, p);
}

extern "C" void kernel_launch(void* const* d_in, const int* in_sizes, int n_in,
                              void* d_out, int out_size, void* d_ws, size_t ws_size,
                              hipStream_t stream) {
  (void)in_sizes; (void)n_in; (void)out_size; (void)ws_size;
  const float* x        = (const float*)d_in[0];
  const float* arch     = (const float*)d_in[1];
  const float* enorm    = (const float*)d_in[2];
  const int*   idx      = (const int*)d_in[3];
  const float* w3       = (const float*)d_in[4];
  const float* w5       = (const float*)d_in[5];
  const float* w7       = (const float*)d_in[6];
  const float* w1       = (const float*)d_in[7];
  const float* sep3_dw  = (const float*)d_in[8];
  const float* sep3_pw  = (const float*)d_in[9];
  const float* sep5_dw  = (const float*)d_in[10];
  const float* sep5_pw  = (const float*)d_in[11];
  const float* dil3     = (const float*)d_in[12];
  const float* dil5     = (const float*)d_in[13];

  char* ws = (char*)d_ws;
  size_t off = 0;
  float* sw = (float*)(ws + off); off += 1024;                    // 10x12 softmax weights
  float* st1 = (float*)(ws + off); off += STATE_BYTES;
  float* st2 = (float*)(ws + off); off += STATE_BYTES;
  float* st3 = (float*)(ws + off); off += STATE_BYTES;
  __bf16* xsb  = (__bf16*)(ws + off); off += (size_t)PCH*NPIX*2;  // gathered bf16 activations
  __bf16* dwb  = (__bf16*)(ws + off); off += (size_t)PCH*NPIX*2;  // depthwise intermediate
  float*  p    = (float*)(ws + off);  off += (size_t)PCH*NPIX*4;  // per-edge mixture accum
  __bf16* wbuf = (__bf16*)(ws + off); off += (size_t)PCH*PCH*49*2; // bf16 weight staging

  const float* states[4] = { x, st1, st2, st3 };
  float* accs[4] = { st1, st2, st3, (float*)d_out };

  softmax_kernel<<<1, 32, 0, stream>>>(arch, enorm, sw);

  const int gP  = (PCH*NPIX + 255)/256;       // per-channel-pixel grids
  const int gS  = (STATE_ELEMS + 255)/256;    // full-state grids

  int e = 0;
  for (int node = 0; node < 4; ++node) {
    float* acc = accs[node];
    zero_kernel<<<gS, 256, 0, stream>>>(acc, STATE_ELEMS);
    for (int i = 0; i <= node; ++i, ++e) {
      const float* s   = states[i];
      const int*   ids = idx + e*PCH;
      const float* swe = sw + e*12;

      gather_kernel<<<gP, 256, 0, stream>>>(s, ids, swe, xsb, p);

      // dense convs: (weights src, K, DIL, mixture index)
      struct { const float* w; int K, DIL, j; size_t elems; } dense[6] = {
        { w3  + (size_t)e*PCH*PCH*9 , 3, 1, 1 , (size_t)PCH*PCH*9  },
        { w5  + (size_t)e*PCH*PCH*25, 5, 1, 2 , (size_t)PCH*PCH*25 },
        { w7  + (size_t)e*PCH*PCH*49, 7, 1, 3 , (size_t)PCH*PCH*49 },
        { w1  + (size_t)e*PCH*PCH   , 1, 1, 4 , (size_t)PCH*PCH    },
        { dil3+ (size_t)e*PCH*PCH*9 , 3, 2, 9 , (size_t)PCH*PCH*9  },
        { dil5+ (size_t)e*PCH*PCH*25, 5, 2, 10, (size_t)PCH*PCH*25 },
      };
      for (int d = 0; d < 6; ++d) {
        int cnt = (int)dense[d].elems;
        cvt_bf16_kernel<<<(cnt+255)/256, 256, 0, stream>>>(dense[d].w, wbuf, cnt);
        run_conv_helper(dense[d].K, dense[d].DIL, xsb, wbuf, swe + dense[d].j, p, stream);
      }

      // separable 3x3: depthwise (VALU) then pointwise (WMMA), weight w[7]
      dw_kernel<3><<<gP, 256, 0, stream>>>(xsb, sep3_dw + (size_t)e*PCH*9, dwb);
      cvt_bf16_kernel<<<(PCH*PCH+255)/256, 256, 0, stream>>>(sep3_pw + (size_t)e*PCH*PCH, wbuf, PCH*PCH);
      run_conv_helper(1, 1, dwb, wbuf, swe + 7, p, stream);

      // separable 5x5, weight w[8]
      dw_kernel<5><<<gP, 256, 0, stream>>>(xsb, sep5_dw + (size_t)e*PCH*25, dwb);
      cvt_bf16_kernel<<<(PCH*PCH+255)/256, 256, 0, stream>>>(sep5_pw + (size_t)e*PCH*PCH, wbuf, PCH*PCH);
      run_conv_helper(1, 1, dwb, wbuf, swe + 8, p, stream);

      // pools on the full state, then scatter the gathered mixture back
      pool_kernel<<<gS, 256, 0, stream>>>(s, swe, acc);
      scatter_kernel<<<gP, 256, 0, stream>>>(p, ids, acc);
    }
  }
}
